// OneHotExclusiveConv_34857954574526
// MI455X (gfx1250) — matrix operand order repaired and checked by reading, not compiled, not run on hardware
//
#include <hip/hip_runtime.h>
#include <hip/hip_bf16.h>

// ---------------------------------------------------------------------------
// OneHotExclusiveConv for MI455X (gfx1250, wave32, WMMA + async LDS staging)
//
// Phase 1: zero feats[N_OUT][160] (f32) in workspace
// Phase 2: pack kernel[160][64] -> f16 in WMMA B-fragment lane layout
// Phase 3: scatter exp(-rate[c]*(t_out[seg]-dt)) into feats (atomic f32)
// Phase 4: WMMA GEMM  out[N_OUT][64] = feats @ W + bias
//          A tile (64 rows x 640 B, contiguous) and packed B (20 KB) are
//          staged into LDS with global_load_async_to_lds_b128 (ASYNCcnt),
//          then 5 x 4 v_wmma_f32_16x16x32_f16 per wave from LDS.
// ---------------------------------------------------------------------------

typedef __attribute__((ext_vector_type(16))) _Float16 v16h;
typedef __attribute__((ext_vector_type(8)))  float    v8f;
typedef __attribute__((ext_vector_type(4)))  float    v4f;

#define C_IN   32
#define KSZ    5
#define KRED   (KSZ * C_IN)   // 160, reduction length
#define F_OUT  64
#define ROWB   640            // bytes per feats row (KRED * 4)
#define LDSP   164            // LDS row pitch in dwords (656 B): 164*4 % 256B
                              // -> per-lane bank-group starts all distinct

// ---------------------------------------------------------------------------
// CDNA5 async global->LDS helpers (inline asm: portable across toolchains)
// ---------------------------------------------------------------------------
__device__ __forceinline__ void async_copy_b128(unsigned lds_byte_addr,
                                                const void* gaddr) {
    asm volatile("global_load_async_to_lds_b128 %0, %1, off"
                 :: "v"(lds_byte_addr), "v"(gaddr) : "memory");
}

__device__ __forceinline__ void wait_async0() {
#if __has_builtin(__builtin_amdgcn_s_wait_asynccnt)
    __builtin_amdgcn_s_wait_asynccnt(0);
#else
    asm volatile("s_wait_asynccnt 0x0" ::: "memory");
#endif
}

// ---------------------------------------------------------------------------
// Phase 1: zero the feats accumulator (grid-stride float4 stores)
// ---------------------------------------------------------------------------
__global__ void zero_feats_kernel(float4* __restrict__ p, int n4) {
    int i = blockIdx.x * blockDim.x + threadIdx.x;
    int stride = gridDim.x * blockDim.x;
    float4 z = make_float4(0.f, 0.f, 0.f, 0.f);
    for (; i < n4; i += stride) p[i] = z;
}

// ---------------------------------------------------------------------------
// Phase 2: pack kernel (f32 [KRED][F_OUT]) into f16 B-fragments.
// Layout: Bpack[kci][nt][half][n][e]  (5*4*2*16*16 f16 = 20 KB)
//   reduction index k = kci*32 + (e>>3)*16 + half*8 + (e&7),  column f = nt*16 + n
// so GEMM lane (n = lane&15, half = lane>>4) reads 16 contiguous f16 (32 B).
// ---------------------------------------------------------------------------
__global__ void pack_kernel_f16(const float* __restrict__ kern,
                                _Float16* __restrict__ bpack, int total) {
    int d = blockIdx.x * blockDim.x + threadIdx.x;
    if (d >= total) return;
    int e    = d & 15;
    int n    = (d >> 4) & 15;
    int half = (d >> 8) & 1;
    int nt   = (d >> 9) & 3;
    int kci  = d >> 11;
    int kred = kci * 32 + (e >> 3) * 16 + half * 8 + (e & 7);
    int f    = nt * 16 + n;
    bpack[d] = (_Float16)kern[kred * F_OUT + f];
}

// ---------------------------------------------------------------------------
// Phase 3: scatter decayed one-hot values into feats.
// rate = softplus(decay_rate) staged in LDS (32 entries).
// ---------------------------------------------------------------------------
__global__ void scatter_kernel(const float* __restrict__ dt,
                               const float* __restrict__ times_out,
                               const int*   __restrict__ skc_arr,
                               const int*   __restrict__ seg_arr,
                               const float* __restrict__ decay_rate,
                               float* __restrict__ feats, int E) {
    __shared__ float s_rate[C_IN];
    if (threadIdx.x < C_IN) {
        float x  = decay_rate[threadIdx.x];
        float ax = fabsf(x);
        // numerically stable softplus: max(x,0) + log1p(exp(-|x|))
        s_rate[threadIdx.x] = fmaxf(x, 0.0f) + log1pf(__expf(-ax));
    }
    __syncthreads();
    int i = blockIdx.x * blockDim.x + threadIdx.x;
    int stride = gridDim.x * blockDim.x;
    for (; i < E; i += stride) {
        int   skc   = skc_arr[i];
        int   seg   = seg_arr[i];
        float delta = times_out[seg] - dt[i];
        float r     = s_rate[skc & (C_IN - 1)];
        float v     = __expf(-r * delta);
        atomicAdd(&feats[(size_t)seg * KRED + skc], v);
    }
}

// ---------------------------------------------------------------------------
// Phase 4: WMMA GEMM. 128 threads = 4 waves; each wave owns a 16x64 tile.
// Block covers 64 rows. A tile + packed B staged in LDS via async copies.
// ---------------------------------------------------------------------------
__device__ __forceinline__ v16h cvt_a_frag(const float* __restrict__ pa) {
    // elements 0..7  <- K = half*8 + 0..7       (pa[0..7])
    // elements 8..15 <- K = 16 + half*8 + 0..7  (pa[16..23])
    v4f f0 = *(const v4f*)(pa);
    v4f f1 = *(const v4f*)(pa + 4);
    v4f f2 = *(const v4f*)(pa + 16);
    v4f f3 = *(const v4f*)(pa + 20);
    v16h a;
    a[0]  = (_Float16)f0.x;  a[1]  = (_Float16)f0.y;
    a[2]  = (_Float16)f0.z;  a[3]  = (_Float16)f0.w;
    a[4]  = (_Float16)f1.x;  a[5]  = (_Float16)f1.y;
    a[6]  = (_Float16)f1.z;  a[7]  = (_Float16)f1.w;
    a[8]  = (_Float16)f2.x;  a[9]  = (_Float16)f2.y;
    a[10] = (_Float16)f2.z;  a[11] = (_Float16)f2.w;
    a[12] = (_Float16)f3.x;  a[13] = (_Float16)f3.y;
    a[14] = (_Float16)f3.z;  a[15] = (_Float16)f3.w;
    return a;
}

__global__ void __launch_bounds__(128)
gemm_wmma_kernel(const float*    __restrict__ feats,
                 const _Float16* __restrict__ bpack,
                 const float*    __restrict__ bias,
                 float*          __restrict__ out,
                 int n_out) {
    __shared__ float    lds_a[64 * LDSP];   // 41,984 B (rows padded to 656 B)
    __shared__ _Float16 lds_b[5 * 4 * 2 * 16 * 16];  // 20,480 B

    const int tid  = threadIdx.x;           // 0..127
    const int lane = tid & 31;
    const int wave = tid >> 5;              // 0..3
    const int half = lane >> 4;             // 0 or 1
    const int m    = lane & 15;             // A row in tile / B col in ntile
    const int rowBase = blockIdx.x * 64 + wave * 16;
    if (blockIdx.x * 64 + 64 > n_out) return;   // block-uniform guard

    // ---- stage A: 64 contiguous rows (40 KB) -> padded LDS rows ----
    {
        const char* gbase = (const char*)feats + (size_t)blockIdx.x * 64 * ROWB;
        unsigned    lbase = (unsigned)(size_t)&lds_a[0];
        #pragma unroll
        for (int it = 0; it < 20; ++it) {            // 64 rows * 40 chunks / 128
            int chunk = tid + it * 128;
            int r = chunk / 40;
            int c = chunk - r * 40;                  // 16-byte units within row
            async_copy_b128(lbase + (unsigned)(r * (LDSP * 4) + c * 16),
                            gbase + (size_t)r * ROWB + (size_t)c * 16);
        }
    }
    // ---- stage B: packed fragments (20 KB, contiguous) ----
    {
        const char* gb = (const char*)bpack;
        unsigned    lb = (unsigned)(size_t)&lds_b[0];
        #pragma unroll
        for (int it = 0; it < 10; ++it) {            // 1280 chunks / 128
            int chunk = tid + it * 128;
            async_copy_b128(lb + (unsigned)(chunk * 16), gb + (size_t)chunk * 16);
        }
    }
    wait_async0();
    __syncthreads();

    // ---- compute: 5 k-chunks x 4 n-tiles of v_wmma_f32_16x16x32_f16 ----
    v8f acc0 = {}, acc1 = {}, acc2 = {}, acc3 = {};
    const float* arow = &lds_a[(wave * 16 + m) * LDSP];

    #pragma unroll
    for (int kci = 0; kci < 5; ++kci) {
        v16h a = cvt_a_frag(arow + kci * 32 + half * 8);

        const _Float16* pb = &lds_b[(size_t)(((kci * 4) * 2 + half) * 16 + m) * 16];
        v16h b0 = *(const v16h*)(pb);          // nt=0
        v16h b1 = *(const v16h*)(pb + 512);    // nt=1 (+2*16*16 halves)
        v16h b2 = *(const v16h*)(pb + 1024);   // nt=2
        v16h b3 = *(const v16h*)(pb + 1536);   // nt=3

        acc0 = __builtin_amdgcn_wmma_f32_16x16x32_f16(false, a, false, b0, (short)0, acc0, false, false);
        acc1 = __builtin_amdgcn_wmma_f32_16x16x32_f16(false, a, false, b1, (short)0, acc1, false, false);
        acc2 = __builtin_amdgcn_wmma_f32_16x16x32_f16(false, a, false, b2, (short)0, acc2, false, false);
        acc3 = __builtin_amdgcn_wmma_f32_16x16x32_f16(false, a, false, b3, (short)0, acc3, false, false);
    }

    // ---- store: C/D layout VGPR r -> D[M = r + 8*half, N = lane&15] ----
    const int rbase = rowBase + half * 8;
    #pragma unroll
    for (int nt = 0; nt < 4; ++nt) {
        const int col = nt * 16 + m;
        const float bv = bias[col];
        const v8f acc = (nt == 0) ? acc0 : (nt == 1) ? acc1 : (nt == 2) ? acc2 : acc3;
        #pragma unroll
        for (int r = 0; r < 8; ++r) {
            out[(size_t)(rbase + r) * F_OUT + col] = acc[r] + bv;
        }
    }
}

// ---------------------------------------------------------------------------
// Host launcher
// ---------------------------------------------------------------------------
extern "C" void kernel_launch(void* const* d_in, const int* in_sizes, int n_in,
                              void* d_out, int out_size, void* d_ws, size_t ws_size,
                              hipStream_t stream) {
    const float* dt        = (const float*)d_in[0];
    const float* times_out = (const float*)d_in[1];
    const int*   skc       = (const int*)  d_in[2];
    const int*   seg       = (const int*)  d_in[3];
    const float* decay     = (const float*)d_in[4];
    const float* kern      = (const float*)d_in[5];
    const float* bias      = (const float*)d_in[6];
    float*       out       = (float*)d_out;

    const int E     = in_sizes[0];
    const int N_OUT = in_sizes[1];

    const size_t featsBytes = (size_t)N_OUT * KRED * sizeof(float);   // ~167 MB
    const int    bpackElems = 5 * 4 * 2 * 16 * 16;                    // 10240 f16
    if (ws_size < featsBytes + (size_t)bpackElems * sizeof(_Float16)) return;

    float*    feats = (float*)d_ws;
    _Float16* bpack = (_Float16*)((char*)d_ws + featsBytes);

    // Phase 1: zero feats
    {
        int n4 = (int)((size_t)N_OUT * KRED / 4);
        zero_feats_kernel<<<4096, 256, 0, stream>>>((float4*)feats, n4);
    }
    // Phase 2: pack kernel -> f16 B fragments
    {
        int blocks = (bpackElems + 255) / 256;
        pack_kernel_f16<<<blocks, 256, 0, stream>>>(kern, bpack, bpackElems);
    }
    // Phase 3: scatter
    {
        int blocks = (E + 255) / 256;
        if (blocks > 8192) blocks = 8192;
        scatter_kernel<<<blocks, 256, 0, stream>>>(dt, times_out, skc, seg, decay, feats, E);
    }
    // Phase 4: WMMA GEMM (64 rows per 128-thread block)
    {
        int blocks = (N_OUT + 63) / 64;
        gemm_wmma_kernel<<<blocks, 128, 0, stream>>>(feats, bpack, bias, out, N_OUT);
    }
}
